// MergeZoom_40235253629065
// MI455X (gfx1250) — compile-verified, matrix operand order checked by reading, not run on hardware
//
#include <hip/hip_runtime.h>
#include <hip/hip_bf16.h>

typedef __attribute__((ext_vector_type(16))) _Float16 v16h;
typedef __attribute__((ext_vector_type(8)))  float    v8f;

namespace {
constexpr int   Bn = 32;
constexpr int   H  = 512;
constexpr int   W  = 512;
constexpr float THRESH = 0.5f;
}

__device__ __forceinline__ int imax(int a, int b) { return a > b ? a : b; }
__device__ __forceinline__ int imin(int a, int b) { return a < b ? a : b; }

// ---------------------------------------------------------------------------
// Row-any flags via WMMA: one wave per 16-row x 512-col strip.
// A[m][k] = (mask[rowbase+m][col] >= 0.5) in f16, B = ones  ->  D[m][*] = rowsum.
// Lane L owns row M = L%16; lane half selects which 16 of the 32 tile columns
// it supplies (K order within a lane is irrelevant since B is all-ones).
// ---------------------------------------------------------------------------
__global__ __launch_bounds__(32) void mz_rowflags(const float* __restrict__ mask,
                                                  unsigned char* __restrict__ rowflag) {
  const int b       = blockIdx.y;
  const int rowbase = blockIdx.x * 16;
  const int lane    = threadIdx.x;   // 0..31, one full wave
  const int m       = lane & 15;
  const int half    = lane >> 4;
  const float* __restrict__ rowp = mask + ((size_t)b * H + (size_t)(rowbase + m)) * W;

  const _Float16 one  = (_Float16)1.0f;
  const _Float16 zero = (_Float16)0.0f;
  v16h ones;
#pragma unroll
  for (int i = 0; i < 16; ++i) ones[i] = one;

  v8f acc = {};
  for (int t = 0; t < 16; ++t) {               // 16 tiles of K=32 cover 512 cols
    const int colbase = t * 32 + half * 16;
    const float4* __restrict__ p = (const float4*)(rowp + colbase);
    v16h a;
#pragma unroll
    for (int q = 0; q < 4; ++q) {
      float4 v = p[q];
      a[q * 4 + 0] = (v.x >= THRESH) ? one : zero;
      a[q * 4 + 1] = (v.y >= THRESH) ? one : zero;
      a[q * 4 + 2] = (v.z >= THRESH) ? one : zero;
      a[q * 4 + 3] = (v.w >= THRESH) ? one : zero;
    }
    acc = __builtin_amdgcn_wmma_f32_16x16x32_f16(false, a, false, ones,
                                                 (short)0, acc, false, false);
  }
  // C/D layout: lane 0 holds (M=0..7, N=0) in acc[0..7]; lane 16 holds M=8..15.
  if ((lane & 15) == 0) {
    const int mofs = half * 8;
#pragma unroll
    for (int j = 0; j < 8; ++j)
      rowflag[(size_t)b * H + (size_t)(rowbase + mofs + j)] = (acc[j] > 0.0f) ? 1 : 0;
  }
}

// ---------------------------------------------------------------------------
// Col-any flags via WMMA: one wave per 16-col x 512-row strip.
// A = ones, B[k][n] = (mask[row][colbase+n] >= 0.5)  ->  D[*][n] = colsum.
// Lane L owns column N = L%16; lane half selects which 16 of the 32 tile rows.
// Lanes 0..15 read 16 consecutive floats per step (coalesced 64B segment).
// ---------------------------------------------------------------------------
__global__ __launch_bounds__(32) void mz_colflags(const float* __restrict__ mask,
                                                  unsigned char* __restrict__ colflag) {
  const int b       = blockIdx.y;
  const int colbase = blockIdx.x * 16;
  const int lane    = threadIdx.x;
  const int n       = lane & 15;
  const int half    = lane >> 4;
  const float* __restrict__ colp = mask + (size_t)b * H * W + (size_t)(colbase + n);

  const _Float16 one  = (_Float16)1.0f;
  const _Float16 zero = (_Float16)0.0f;
  v16h ones;
#pragma unroll
  for (int i = 0; i < 16; ++i) ones[i] = one;

  v8f acc = {};
  for (int t = 0; t < 16; ++t) {               // 16 tiles of K=32 cover 512 rows
    const int rowb = t * 32 + half * 16;
    v16h bm;
#pragma unroll
    for (int j = 0; j < 16; ++j) {
      float v = colp[(size_t)(rowb + j) * W];
      bm[j] = (v >= THRESH) ? one : zero;
    }
    acc = __builtin_amdgcn_wmma_f32_16x16x32_f16(false, ones, false, bm,
                                                 (short)0, acc, false, false);
  }
  // Lane L (0..15) holds (M=0, N=L) in acc[0] -> column sum for colbase+L.
  if (lane < 16)
    colflag[(size_t)b * W + (size_t)(colbase + n)] = (acc[0] > 0.0f) ? 1 : 0;
}

// ---------------------------------------------------------------------------
// BBox per batch, replicating argmax semantics (no-hit -> first=0, last=L-1).
// ---------------------------------------------------------------------------
__global__ __launch_bounds__(32) void mz_bbox(const unsigned char* __restrict__ rowflag,
                                              const unsigned char* __restrict__ colflag,
                                              int* __restrict__ bbox) {
  const int b = threadIdx.x;
  if (b >= Bn) return;

  const unsigned char* rf = rowflag + (size_t)b * H;
  int first = 0;
  for (int i = 0; i < H; ++i) { if (rf[i]) { first = i; break; } }
  int last = H - 1;
  for (int i = H - 1; i >= 0; --i) { if (rf[i]) { last = i; break; } }
  bbox[b * 4 + 0] = first - 1;   // xl
  bbox[b * 4 + 1] = last + 1;    // xr

  const unsigned char* cf = colflag + (size_t)b * W;
  first = 0;
  for (int i = 0; i < W; ++i) { if (cf[i]) { first = i; break; } }
  last = W - 1;
  for (int i = W - 1; i >= 0; --i) { if (cf[i]) { last = i; break; } }
  bbox[b * 4 + 2] = first - 1;   // yl
  bbox[b * 4 + 3] = last + 1;    // yr
}

// ---------------------------------------------------------------------------
// Bilinear crop-resize of (mask>=0.5)*image back to HxW. One thread per pixel.
// Mirrors _axis_coords exactly: src clip to [lo,hi-1], i0=max(floor,0),
// i1=min(i0+1,hi-1), w = src - i0 (unclamped, may be negative at the edge).
// ---------------------------------------------------------------------------
__global__ __launch_bounds__(256) void mz_resize(const float* __restrict__ mask,
                                                 const float* __restrict__ image,
                                                 const int* __restrict__ bbox,
                                                 float* __restrict__ out) {
  const int b = blockIdx.z;
  const int y = blockIdx.y;
  const int x = blockIdx.x * 256 + threadIdx.x;

  const int xl = bbox[b * 4 + 0], xr = bbox[b * 4 + 1];
  const int yl = bbox[b * 4 + 2], yr = bbox[b * 4 + 3];

  // Row coords (from row bbox, out_len = H, index = y)
  const float sizeR = (float)(xr - xl);
  float srcR = ((float)y + 0.5f) * sizeR / (float)H - 0.5f + (float)xl;
  srcR = fminf(fmaxf(srcR, (float)xl), (float)(xr - 1));
  const int   r0 = imax((int)floorf(srcR), 0);
  const int   r1 = imin(r0 + 1, xr - 1);
  const float wr = srcR - (float)r0;

  // Col coords (from col bbox, out_len = W, index = x)
  const float sizeC = (float)(yr - yl);
  float srcC = ((float)x + 0.5f) * sizeC / (float)W - 0.5f + (float)yl;
  srcC = fminf(fmaxf(srcC, (float)yl), (float)(yr - 1));
  const int   c0 = imax((int)floorf(srcC), 0);
  const int   c1 = imin(c0 + 1, yr - 1);
  const float wc = srcC - (float)c0;

  const size_t mbase = (size_t)b * H * W;
  const size_t p00 = mbase + (size_t)r0 * W + c0;
  const size_t p01 = mbase + (size_t)r0 * W + c1;
  const size_t p10 = mbase + (size_t)r1 * W + c0;
  const size_t p11 = mbase + (size_t)r1 * W + c1;

  const float m00 = (mask[p00] >= THRESH) ? 1.0f : 0.0f;
  const float m01 = (mask[p01] >= THRESH) ? 1.0f : 0.0f;
  const float m10 = (mask[p10] >= THRESH) ? 1.0f : 0.0f;
  const float m11 = (mask[p11] >= THRESH) ? 1.0f : 0.0f;

  const size_t obase = (mbase + (size_t)y * W + x) * 3;
#pragma unroll
  for (int ch = 0; ch < 3; ++ch) {
    const float v00 = m00 * image[p00 * 3 + ch];
    const float v01 = m01 * image[p01 * 3 + ch];
    const float v10 = m10 * image[p10 * 3 + ch];
    const float v11 = m11 * image[p11 * 3 + ch];
    const float top = v00 * (1.0f - wc) + v01 * wc;
    const float bot = v10 * (1.0f - wc) + v11 * wc;
    out[obase + ch] = top * (1.0f - wr) + bot * wr;
  }
}

// ---------------------------------------------------------------------------
extern "C" void kernel_launch(void* const* d_in, const int* in_sizes, int n_in,
                              void* d_out, int out_size, void* d_ws, size_t ws_size,
                              hipStream_t stream) {
  const float* mask  = (const float*)d_in[0];  // (32,512,512,1) f32
  const float* image = (const float*)d_in[1];  // (32,512,512,3) f32
  float*       out   = (float*)d_out;          // (32,512,512,3) f32

  // Workspace layout (fully rewritten every call; no reliance on contents):
  //   [0,                Bn*H)            row flags (u8)
  //   [Bn*H,             Bn*H + Bn*W)     col flags (u8)
  //   [Bn*H + Bn*W, ...]                  bbox (Bn * 4 ints, 4B-aligned: 32768)
  unsigned char* rowflag = (unsigned char*)d_ws;
  unsigned char* colflag = rowflag + (size_t)Bn * H;
  int*           bbox    = (int*)(colflag + (size_t)Bn * W);

  mz_rowflags<<<dim3(H / 16, Bn), dim3(32), 0, stream>>>(mask, rowflag);
  mz_colflags<<<dim3(W / 16, Bn), dim3(32), 0, stream>>>(mask, colflag);
  mz_bbox<<<dim3(1), dim3(32), 0, stream>>>(rowflag, colflag, bbox);
  mz_resize<<<dim3(W / 256, H, Bn), dim3(256), 0, stream>>>(mask, image, bbox, out);
}
